// IntraChrModel_78640851189997
// MI455X (gfx1250) — compile-verified, matrix operand order checked by reading, not run on hardware
//
// MI455X (gfx1250, CDNA5) implementation of the IntraChrModel forward pass.
// GEMM-shaped work -> v_wmma_f32_16x16x32_f16 with LDS tiles staged directly
// in WMMA fragment order (2x ds_load_b128 per fragment, v4f global loads,
// packed v4h LDS stores). Boundary handling is branchless (clamped loads +
// selects) to avoid wave32 exec-mask serialization. The sequential Mamba scan
// double-buffers B/C vectors in LDS via global_load_async_to_lds_b32 +
// s_wait_asynccnt (ASYNCcnt), overlapping fetch with the v_exp/FMA chain.
#include <hip/hip_runtime.h>
#include <hip/hip_bf16.h>
#include <math.h>

typedef __attribute__((ext_vector_type(16))) _Float16 v16h;
typedef __attribute__((ext_vector_type(4)))  _Float16 v4h;
typedef __attribute__((ext_vector_type(8)))  float    v8f;
typedef __attribute__((ext_vector_type(4)))  float    v4f;

#define BATCH 8
#define LSEQ  8192
#define MROWS (BATCH * LSEQ)
#define HDIM  128
#define DIC   256   // mamba inner dim
#define DSN   16    // SSM states

__device__ __forceinline__ int iclamp(int v, int lo, int hi) {
  return v < lo ? lo : (v > hi ? hi : v);
}

// activation codes: 0=none 1=relu 2=silu 3=softplus 4=sigmoid
__device__ __forceinline__ float actf(float x, int act) {
  switch (act) {
    case 1: return x > 0.f ? x : 0.f;
    case 2: return x / (1.f + __expf(-x));
    case 3: return x > 20.f ? x : log1pf(__expf(x));
    case 4: return 1.f / (1.f + __expf(-x));
    default: return x;
  }
}

// (row/col, k) -> (lane, elem) mapping of the 16-bit 16x32 WMMA fragment
// (ISA 7.12.2): lane = idx15 + 16*k[3]; elem = k[2:0] + 8*k[4].
__device__ __forceinline__ int frag_lane(int idx15, int k) {
  return idx15 | (((k >> 3) & 1) << 4);
}
__device__ __forceinline__ int frag_elem(int k) {
  return (k & 7) | (((k >> 4) & 1) << 3);
}

// ---------------------------------------------------------------------------
// Generic WMMA GEMM: C[M,N] = act(A[M,K] * W[K,N] + bias + resid)
// Tile 64x64, K-step 32. 8 waves in 4(M) x 2(N); each wave: two 16x16x32 WMMAs.
// LDS tiles are pre-permuted to fragment order [subtile][lane][16 halves] so a
// fragment fetch is one contiguous 32B load (2x ds_load_b128).
// ---------------------------------------------------------------------------
#define TM 64
#define TN 64
#define TKK 32

__global__ __launch_bounds__(256) void k_gemm_wmma(
    const float* __restrict__ A, int lda,
    const float* __restrict__ W, int ldw,
    const float* __restrict__ bias,
    const float* __restrict__ resid,
    float* __restrict__ C, int ldc,
    int M, int N, int K, int act)
{
  __shared__ __attribute__((aligned(32))) _Float16 AsF[4][32][16];
  __shared__ __attribute__((aligned(32))) _Float16 BsF[4][32][16];
  const int tid  = threadIdx.x;
  const int lane = tid & 31;
  const int wave = tid >> 5;
  const int bm = blockIdx.x * TM;
  const int bn = blockIdx.y * TN;
  const int asub  = wave & 3;         // M subtile (16 rows)
  const int bsub0 = (wave >> 2) * 2;  // first of two N subtiles
  v8f acc0 = {}; v8f acc1 = {};

  for (int k0 = 0; k0 < K; k0 += TKK) {
    const bool fullA = (bm + TM <= M) && (k0 + TKK <= K);
    const bool fullB = (bn + TN <= N) && (k0 + TKK <= K);
    if (fullA) {
      for (int t = tid; t < 512; t += 256) {           // 64 rows x 8 quads
        int r = t >> 3, kq = (t & 7) << 2;
        v4f a = *reinterpret_cast<const v4f*>(&A[(size_t)(bm + r) * lda + k0 + kq]);
        v4h h; h[0] = (_Float16)a[0]; h[1] = (_Float16)a[1];
               h[2] = (_Float16)a[2]; h[3] = (_Float16)a[3];
        *reinterpret_cast<v4h*>(&AsF[r >> 4][frag_lane(r & 15, kq)][frag_elem(kq)]) = h;
      }
      if (k0 + TKK < K)  // warm next strip (global_prefetch_b8)
        __builtin_prefetch(&A[(size_t)(bm + (tid & 63)) * lda + k0 + TKK], 0, 1);
    } else {
      // branchless edge path: clamp index, load, select 0 for OOB lanes
      const int k  = tid & 31;                 // constant per thread
      const int gk = k0 + k;
      const int gkc = iclamp(gk, 0, K - 1);
      const bool kok = gk < K;
      const int ln = frag_lane(0, k), eb = frag_elem(k);
      for (int r = tid >> 5; r < TM; r += 8) {
        int gr  = bm + r;
        int grc = iclamp(gr, 0, M - 1);
        float v = A[(size_t)grc * lda + gkc];
        v = (kok && gr < M) ? v : 0.f;
        AsF[r >> 4][ln | (r & 15)][eb] = (_Float16)v;
      }
    }
    {
      const int col = tid & 63;                // constant per thread
      const int gc  = bn + col;
      const int gcc = iclamp(gc, 0, N - 1);
      const bool cok = gc < N;
      for (int kqi = tid >> 6; kqi < 8; kqi += 4) {
        int kq = kqi << 2;
        v4h h;
        if (fullB) {
#pragma unroll
          for (int j = 0; j < 4; ++j)
            h[j] = (_Float16)W[(size_t)(k0 + kq + j) * ldw + gc];
        } else {
#pragma unroll
          for (int j = 0; j < 4; ++j) {
            int gk = k0 + kq + j;
            float v = W[(size_t)iclamp(gk, 0, K - 1) * ldw + gcc];
            h[j] = (_Float16)((cok && gk < K) ? v : 0.f);
          }
        }
        *reinterpret_cast<v4h*>(&BsF[col >> 4][frag_lane(col & 15, kq)][frag_elem(kq)]) = h;
      }
    }
    __syncthreads();
    v16h af  = *reinterpret_cast<const v16h*>(&AsF[asub][lane][0]);
    v16h bf0 = *reinterpret_cast<const v16h*>(&BsF[bsub0][lane][0]);
    v16h bf1 = *reinterpret_cast<const v16h*>(&BsF[bsub0 + 1][lane][0]);
    acc0 = __builtin_amdgcn_wmma_f32_16x16x32_f16(false, af, false, bf0,
                                                  (short)0, acc0, false, false);
    acc1 = __builtin_amdgcn_wmma_f32_16x16x32_f16(false, af, false, bf1,
                                                  (short)0, acc1, false, false);
    __syncthreads();
  }

  // D layout: VGPR r -> M=r (lanes 0-15) / M=r+8 (lanes 16-31); N = lane&15
  const int gm0 = bm + asub * 16 + ((lane >> 4) ? 8 : 0);
  const int n0  = bn + bsub0 * 16 + (lane & 15);
  const int n1  = n0 + 16;
  float bv0 = (bias && n0 < N) ? bias[n0] : 0.f;
  float bv1 = (bias && n1 < N) ? bias[n1] : 0.f;
#pragma unroll
  for (int r = 0; r < 8; ++r) {
    int gm = gm0 + r;
    if (gm >= M) continue;
    if (n0 < N) {
      float v = acc0[r] + bv0;
      if (resid) v += resid[(size_t)gm * ldc + n0];
      C[(size_t)gm * ldc + n0] = actf(v, act);
    }
    if (n1 < N) {
      float v = acc1[r] + bv1;
      if (resid) v += resid[(size_t)gm * ldc + n1];
      C[(size_t)gm * ldc + n1] = actf(v, act);
    }
  }
}

// ---------------------------------------------------------------------------
// WMMA implicit-GEMM conv (k=3, pad=1) with scale/shift + relu.
// in: [B][Ci][L] channel-major; A-tile row = position, col = ci*3+dk (gathered
// branchless with halo clamps); B-tile contiguous in k -> v4f fast path.
// ---------------------------------------------------------------------------
__global__ __launch_bounds__(256) void k_conv_wmma(
    const float* __restrict__ in, const float* __restrict__ w,  // w: [Co][Ci*3]
    const float* __restrict__ bc, const float* __restrict__ g,
    const float* __restrict__ beta, float* __restrict__ out,
    int Ci, int Co, int tokenOut)
{
  __shared__ __attribute__((aligned(32))) _Float16 AsF[4][32][16];
  __shared__ __attribute__((aligned(32))) _Float16 BsF[4][32][16];
  const int Kc = Ci * 3;              // 192 or 384: multiple of 32
  const int b  = blockIdx.z;
  const int bm = blockIdx.x * TM;     // positions
  const int bn = blockIdx.y * TN;     // out channels (Co=128 -> always full)
  const int tid  = threadIdx.x;
  const int lane = tid & 31;
  const int wave = tid >> 5;
  const int asub  = wave & 3;
  const int bsub0 = (wave >> 2) * 2;
  v8f acc0 = {}; v8f acc1 = {};

  for (int k0 = 0; k0 < Kc; k0 += TKK) {
    {
      // im2col gather: per-thread-invariant (ci,dk,elem); branchless halo
      const int k  = tid & 31;
      const int kk = k0 + k;
      const int ci = kk / 3, dk = kk - ci * 3;
      const float* src = in + ((size_t)b * Ci + ci) * LSEQ;
      const int ln = frag_lane(0, k), eb = frag_elem(k);
      for (int r = tid >> 5; r < TM; r += 8) {
        int p  = bm + r + dk - 1;
        int pc = iclamp(p, 0, LSEQ - 1);
        float v = src[pc];
        v = (p == pc) ? v : 0.f;
        AsF[r >> 4][ln | (r & 15)][eb] = (_Float16)v;
      }
    }
    {
      const int col = tid & 63;
      const float* wr = w + (size_t)(bn + col) * Kc + k0;
      for (int kqi = tid >> 6; kqi < 8; kqi += 4) {
        int kq = kqi << 2;
        v4f a = *reinterpret_cast<const v4f*>(wr + kq);
        v4h h; h[0] = (_Float16)a[0]; h[1] = (_Float16)a[1];
               h[2] = (_Float16)a[2]; h[3] = (_Float16)a[3];
        *reinterpret_cast<v4h*>(&BsF[col >> 4][frag_lane(col & 15, kq)][frag_elem(kq)]) = h;
      }
    }
    __syncthreads();
    v16h af  = *reinterpret_cast<const v16h*>(&AsF[asub][lane][0]);
    v16h bf0 = *reinterpret_cast<const v16h*>(&BsF[bsub0][lane][0]);
    v16h bf1 = *reinterpret_cast<const v16h*>(&BsF[bsub0 + 1][lane][0]);
    acc0 = __builtin_amdgcn_wmma_f32_16x16x32_f16(false, af, false, bf0,
                                                  (short)0, acc0, false, false);
    acc1 = __builtin_amdgcn_wmma_f32_16x16x32_f16(false, af, false, bf1,
                                                  (short)0, acc1, false, false);
    __syncthreads();
  }

  const int gm0 = bm + asub * 16 + ((lane >> 4) ? 8 : 0);
  const int n0  = bn + bsub0 * 16 + (lane & 15);
  const int n1  = n0 + 16;
#pragma unroll
  for (int r = 0; r < 8; ++r) {
    int p = gm0 + r;
#pragma unroll
    for (int half = 0; half < 2; ++half) {
      int co = half ? n1 : n0;
      if (co >= Co) continue;
      float v = (half ? acc1[r] : acc0[r]) + bc[co];
      v = g[co] * v + beta[co];
      v = fmaxf(v, 0.f);
      if (tokenOut) out[((size_t)b * LSEQ + p) * Co + co] = v;
      else          out[((size_t)b * Co + co) * LSEQ + p] = v;
    }
  }
}

// first conv layer (ci=1 -> 64): plain VALU, branchless halo
__global__ void k_conv1(const float* __restrict__ x, const float* __restrict__ w,
                        const float* __restrict__ bc, const float* __restrict__ g,
                        const float* __restrict__ beta, float* __restrict__ out)
{
  size_t idx = (size_t)blockIdx.x * blockDim.x + threadIdx.x;
  if (idx >= (size_t)BATCH * 64 * LSEQ) return;
  int p  = (int)(idx % LSEQ);
  int co = (int)((idx / LSEQ) & 63);
  int b  = (int)(idx / ((size_t)64 * LSEQ));
  const float* xb = x + (size_t)b * LSEQ;
  float s = bc[co];
#pragma unroll
  for (int dk = 0; dk < 3; ++dk) {
    int pp = p + dk - 1;
    int pc = iclamp(pp, 0, LSEQ - 1);
    float v = xb[pc];
    s += (pp == pc) ? w[co * 3 + dk] * v : 0.f;
  }
  s = g[co] * s + beta[co];
  out[((size_t)b * 64 + co) * LSEQ + p] = fmaxf(s, 0.f);
}

__global__ void k_embed(const float* __restrict__ x, const float* __restrict__ we,
                        const float* __restrict__ be, float* __restrict__ out)
{
  size_t idx = (size_t)blockIdx.x * blockDim.x + threadIdx.x;
  if (idx >= (size_t)MROWS * HDIM) return;
  int c = (int)(idx & 127);
  size_t m = idx >> 7;
  out[idx] = x[m] * we[c] + be[c];
}

__global__ void k_posenc(float* __restrict__ xm)
{
  size_t idx = (size_t)blockIdx.x * blockDim.x + threadIdx.x;
  if (idx >= (size_t)MROWS * HDIM) return;
  int c = (int)(idx & 127);
  size_t m = idx >> 7;
  int p = (int)(m & (LSEQ - 1));
  int i2 = (c >> 1) * 2;
  float div = __expf(-(float)i2 * (logf(10000.f) / (float)HDIM));
  float ang = (float)p * div;
  xm[idx] += (c & 1) ? __cosf(ang) : __sinf(ang);
}

__global__ void k_layernorm(const float* __restrict__ in, const float* __restrict__ g,
                            const float* __restrict__ b, float* __restrict__ out)
{
  size_t m = (size_t)blockIdx.x * blockDim.x + threadIdx.x;
  if (m >= MROWS) return;
  const float* r = in + m * HDIM;
  float mu = 0.f;
  for (int c = 0; c < HDIM; ++c) mu += r[c];
  mu *= (1.f / HDIM);
  float v = 0.f;
  for (int c = 0; c < HDIM; ++c) { float d = r[c] - mu; v += d * d; }
  v *= (1.f / HDIM);
  float is = rsqrtf(v + 1e-5f);
  float* o = out + m * HDIM;
  for (int c = 0; c < HDIM; ++c) o[c] = (r[c] - mu) * is * g[c] + b[c];
}

// depthwise causal conv K=4 + SiLU; dir=0 past taps, dir=1 future taps
__global__ void k_dwconv(const float* __restrict__ xz, const float* __restrict__ cw,
                         const float* __restrict__ cb, float* __restrict__ xa, int dir)
{
  size_t idx = (size_t)blockIdx.x * blockDim.x + threadIdx.x;
  if (idx >= (size_t)MROWS * DIC) return;
  int d = (int)(idx & 255);
  size_t m = idx >> 8;
  int p = (int)(m & (LSEQ - 1));
  size_t b = m >> 13;
  const float* base = xz + (size_t)b * LSEQ * 512 + d;
  float s = cb[d];
#pragma unroll
  for (int k = 0; k < 4; ++k) {
    int pp = dir ? (p + 3 - k) : (p - 3 + k);
    int pc = iclamp(pp, 0, LSEQ - 1);
    float v = base[(size_t)pc * 512];
    s += (pp == pc) ? cw[d * 4 + k] * v : 0.f;
  }
  xa[idx] = s / (1.f + __expf(-s));  // silu
}

// Sequential SSM scan: one block per batch; 256 threads = 256 channels; 16
// states/channel in VGPRs. Per-step B/C vectors (32 floats) are double-
// buffered in LDS and prefetched one step ahead with the CDNA5 async
// global->LDS path (ASYNCcnt), overlapping the fetch with the v_exp/FMA chain.
__global__ __launch_bounds__(256) void k_scan(
    const float* __restrict__ dt, const float* __restrict__ xa,
    const float* __restrict__ dbc, const float* __restrict__ alog,
    float* __restrict__ ys, int dir)
{
  __shared__ __attribute__((aligned(16))) float sBC[2][32];
  const int b = blockIdx.x;
  const int d = threadIdx.x;
  float An[DSN];
#pragma unroll
  for (int s = 0; s < DSN; ++s) An[s] = -__expf(alog[d * DSN + s]);
  float h[DSN];
#pragma unroll
  for (int s = 0; s < DSN; ++s) h[s] = 0.f;

  // preload step 0 (wave 0: one lane per element)
  {
    int p0 = dir ? (LSEQ - 1) : 0;
    size_t m0 = (size_t)b * LSEQ + p0;
    if (threadIdx.x < 32) {
      const float* gp = dbc + m0 * 40 + 8 + threadIdx.x;
      unsigned lo = (unsigned)(uintptr_t)&sBC[0][threadIdx.x];
      asm volatile("global_load_async_to_lds_b32 %0, %1, off"
                   :: "v"(lo), "v"(gp) : "memory");
    }
    asm volatile("s_wait_asynccnt 0x0" ::: "memory");
    __syncthreads();
  }

  for (int t = 0; t < LSEQ; ++t) {
    int p = dir ? (LSEQ - 1 - t) : t;
    size_t m = (size_t)b * LSEQ + p;
    int cur = t & 1;
    if ((t + 1 < LSEQ) && threadIdx.x < 32) {   // async prefetch of step t+1
      int pn = dir ? (LSEQ - 2 - t) : (t + 1);
      size_t mn = (size_t)b * LSEQ + pn;
      const float* gp = dbc + mn * 40 + 8 + threadIdx.x;
      unsigned lo = (unsigned)(uintptr_t)&sBC[cur ^ 1][threadIdx.x];
      asm volatile("global_load_async_to_lds_b32 %0, %1, off"
                   :: "v"(lo), "v"(gp) : "memory");
    }
    float dtv = dt[m * DIC + d];
    float dx  = dtv * xa[m * DIC + d];
    const float* sB = &sBC[cur][0];
    const float* sC = &sBC[cur][16];
    float y = 0.f;
#pragma unroll
    for (int s = 0; s < DSN; ++s) {
      h[s] = __expf(dtv * An[s]) * h[s] + dx * sB[s];
      y += h[s] * sC[s];
    }
    ys[m * DIC + d] = y;
    asm volatile("s_wait_asynccnt 0x0" ::: "memory");
    __syncthreads();
  }
}

// y = (ys + D*xa) * silu(z), in place over ys
__global__ void k_ymix(float* __restrict__ ys, const float* __restrict__ xa,
                       const float* __restrict__ xz, const float* __restrict__ Dp)
{
  size_t idx = (size_t)blockIdx.x * blockDim.x + threadIdx.x;
  if (idx >= (size_t)MROWS * DIC) return;
  int d = (int)(idx & 255);
  size_t m = idx >> 8;
  float z = xz[m * 512 + 256 + d];
  ys[idx] = (ys[idx] + Dp[d] * xa[idx]) * (z / (1.f + __expf(-z)));
}

// column mean over L for token-major [B][L][128] -> [B][128]
__global__ __launch_bounds__(256) void k_colmean(const float* __restrict__ in,
                                                 float* __restrict__ out)
{
  __shared__ float red[256];
  int b = blockIdx.x >> 7, c = blockIdx.x & 127;
  float s = 0.f;
  for (int t = threadIdx.x; t < LSEQ; t += 256)
    s += in[((size_t)b * LSEQ + t) * HDIM + c];
  red[threadIdx.x] = s; __syncthreads();
  for (int o = 128; o > 0; o >>= 1) {
    if ((int)threadIdx.x < o) red[threadIdx.x] += red[threadIdx.x + o];
    __syncthreads();
  }
  if (threadIdx.x == 0) out[b * HDIM + c] = red[0] / (float)LSEQ;
}

// x_main_weighted[b,k] = sum_l mef16[b,l,k] * sigmoid(mew[l])
__global__ __launch_bounds__(256) void k_wmean(const float* __restrict__ mef16,
                                               const float* __restrict__ mew,
                                               float* __restrict__ xmw)
{
  __shared__ float red[256];
  int b = blockIdx.x >> 4, k = blockIdx.x & 15;
  float s = 0.f;
  for (int l = threadIdx.x; l < LSEQ; l += 256) {
    float wv = 1.f / (1.f + __expf(-mew[l]));
    s += mef16[((size_t)b * LSEQ + l) * 16 + k] * wv;
  }
  red[threadIdx.x] = s; __syncthreads();
  for (int o = 128; o > 0; o >>= 1) {
    if ((int)threadIdx.x < o) red[threadIdx.x] += red[threadIdx.x + o];
    __syncthreads();
  }
  if (threadIdx.x == 0) xmw[b * 16 + k] = red[0];
}

// top-200 of sigmoid(mew) by rank counting (ties -> lower index, = lax.top_k);
// also emits sigmoid(mew) as the main_weights output.
__global__ __launch_bounds__(256) void k_top200(const float* __restrict__ mew,
                                                int* __restrict__ topi,
                                                float* __restrict__ mwout)
{
  __shared__ float sw[LSEQ];  // 32 KB
  for (int i = threadIdx.x; i < LSEQ; i += 256)
    sw[i] = 1.f / (1.f + __expf(-mew[i]));
  __syncthreads();
  int l = blockIdx.x * 256 + threadIdx.x;
  float w = sw[l];
  mwout[l] = w;
  int rank = 0;
  for (int j = 0; j < LSEQ; ++j) {
    float wj = sw[j];
    rank += (wj > w) || (wj == w && j < l);
  }
  if (rank < 200) topi[rank] = l;
}

__global__ __launch_bounds__(256) void k_pairs(
    const float* __restrict__ xf, const int* __restrict__ topi,
    const int* __restrict__ seli, const int* __restrict__ pj,
    const float* __restrict__ wp, const float* __restrict__ bp,
    float* __restrict__ scores, int* __restrict__ iidx)
{
  __shared__ float red[256];
  __shared__ float sb[BATCH];
  int p  = blockIdx.x;
  int ii = topi[seli[p]];
  int jj = pj[p];
  int b = threadIdx.x >> 5, lane = threadIdx.x & 31;
  float s = 0.f;
  for (int c = lane; c < HDIM; c += 32) {
    s += xf[((size_t)b * LSEQ + ii) * HDIM + c] * wp[c];
    s += xf[((size_t)b * LSEQ + jj) * HDIM + c] * wp[HDIM + c];
  }
  red[threadIdx.x] = s; __syncthreads();
  if (lane == 0) {
    float t = 0.f;
    for (int i = 0; i < 32; ++i) t += red[b * 32 + i];
    t += bp[0];
    sb[b] = 1.f / (1.f + __expf(-t));
  }
  __syncthreads();
  if (threadIdx.x == 0) {
    float t = 0.f;
    for (int i = 0; i < BATCH; ++i) t += sb[i];
    scores[p] = t / (float)BATCH;
    iidx[p] = ii;
  }
}

__global__ void k_rank200(const float* __restrict__ scores, const int* __restrict__ iidx,
                          const int* __restrict__ pj, float* __restrict__ outScores,
                          int* __restrict__ outPairs)
{
  int p = threadIdx.x;
  if (p >= 200) return;
  float s = scores[p];
  int rank = 0;
  for (int q = 0; q < 200; ++q) {
    float sq = scores[q];
    rank += (sq > s) || (sq == s && q < p);
  }
  outScores[rank] = s;
  outPairs[2 * rank + 0] = iidx[p];
  outPairs[2 * rank + 1] = pj[p];
}

// final head: x_main = relu(xmw@W_agg+b), feat=[cmean|fmean|x_main] (272),
// e1(relu) e2(relu) e3 o1(relu) o2 -> pred[b]. Tiny; one thread per batch row.
__global__ void k_head(const float* __restrict__ cmean, const float* __restrict__ fmean,
                       const float* __restrict__ xmw,
                       const float* __restrict__ Wagg, const float* __restrict__ bagg,
                       const float* __restrict__ We1, const float* __restrict__ be1,
                       const float* __restrict__ We2, const float* __restrict__ be2,
                       const float* __restrict__ We3, const float* __restrict__ be3,
                       const float* __restrict__ Wo1, const float* __restrict__ bo1,
                       const float* __restrict__ Wo2, const float* __restrict__ bo2,
                       float* __restrict__ pred)
{
  int b = threadIdx.x;
  if (b >= BATCH) return;
  float feat[272];
  for (int j = 0; j < 16; ++j) {
    float s = bagg[j];
    for (int k = 0; k < 16; ++k) s += xmw[b * 16 + k] * Wagg[k * 16 + j];
    feat[256 + j] = fmaxf(s, 0.f);
  }
  for (int c = 0; c < 128; ++c) { feat[c] = cmean[b * 128 + c]; feat[128 + c] = fmean[b * 128 + c]; }
  float e1[128];
  for (int j = 0; j < 128; ++j) {
    float s = be1[j];
    for (int i = 0; i < 272; ++i) s += feat[i] * We1[i * 128 + j];
    e1[j] = fmaxf(s, 0.f);
  }
  float e2[64];
  for (int j = 0; j < 64; ++j) {
    float s = be2[j];
    for (int i = 0; i < 128; ++i) s += e1[i] * We2[i * 64 + j];
    e2[j] = fmaxf(s, 0.f);
  }
  float e3[32];
  for (int j = 0; j < 32; ++j) {
    float s = be3[j];
    for (int i = 0; i < 64; ++i) s += e2[i] * We3[i * 32 + j];
    e3[j] = s;
  }
  float o1[16];
  for (int j = 0; j < 16; ++j) {
    float s = bo1[j];
    for (int i = 0; i < 32; ++i) s += e3[i] * Wo1[i * 16 + j];
    o1[j] = fmaxf(s, 0.f);
  }
  float pr = bo2[0];
  for (int i = 0; i < 16; ++i) pr += o1[i] * Wo2[i];
  pred[b] = pr;
}

// ---------------------------------------------------------------------------
static void gemm(const float* A, int lda, const float* W, int ldw,
                 const float* bias, const float* resid, float* C, int ldc,
                 int M, int N, int K, int act, hipStream_t s)
{
  dim3 g((M + TM - 1) / TM, (N + TN - 1) / TN);
  k_gemm_wmma<<<g, 256, 0, s>>>(A, lda, W, ldw, bias, resid, C, ldc, M, N, K, act);
}

extern "C" void kernel_launch(void* const* d_in, const int* in_sizes, int n_in,
                              void* d_out, int out_size, void* d_ws, size_t ws_size,
                              hipStream_t stream)
{
  (void)in_sizes; (void)n_in; (void)out_size; (void)ws_size;
  // Flattened input indices (setup_inputs insertion order, leaves in order):
  // 0:x 1:W_embed 2:b_embed 3..18:cnn{w,b,g,beta}x4 19:W_proj 20:b_proj
  // 21..44:bimamba0{ln_g,ln_b,fwd[11],bwd[11]} 45..68:bimamba1
  // 69:W_me1 70:b_me1 71:W_me2 72:b_me2 73:W_agg 74:b_agg 75:W_pair 76:b_pair
  // 77:W_e1 78:b_e1 79:W_e2 80:b_e2 81:W_e3 82:b_e3 83:W_o1 84:b_o1
  // 85:W_o2 86:b_o2 87:main_effect_weights 88:pair_sel_i 89:pair_j
  auto F = [&](int i) { return (const float*)d_in[i]; };
  const float* x   = F(0);
  const int* seli  = (const int*)d_in[88];
  const int* pj    = (const int*)d_in[89];
  float* out = (float*)d_out;
  float* Wws = (float*)d_ws;
  const size_t Msz = (size_t)MROWS;

  // workspace layout (floats), with aliasing of dead regions
  float* XM   = Wws;                    // x_mamba / x_final   [M,128]
  float* XN   = XM + Msz * 128;         // layernorm out       [M,128]
  float* XC   = XN + Msz * 128;         // cnn out token-major [M,128]
  float* XZ   = XC + Msz * 128;         // in-proj out         [M,512]
  float* XEMB = XZ;                     // alias: embed (dead before XZ used)
  float* MEF  = XZ;                     // alias: me1 out      [M,64]
  float* MEF16= XZ + Msz * 64;          // me2 out             [M,16]
  float* XA   = XZ + Msz * 512;         // conv+silu out       [M,256]
  float* DTb  = XA + Msz * 256;         // dt (softplus)       [M,256]
  float* CP0  = DTb;                    // alias: cnn ping     [B,128,L]
  float* YSb  = DTb + Msz * 256;        // scan out / y        [M,256]
  float* CP1  = YSb;                    // alias: cnn pong
  float* DBC  = YSb + Msz * 256;        // x-proj out          [M,40]
  float* SM     = DBC + Msz * 48;
  float* CMEAN  = SM;                   // [8,128]
  float* FMEAN  = SM + 1024;            // [8,128]
  float* XMW    = SM + 2048;            // [8,16]
  float* SCORES = SM + 2176;            // [200]
  int*   TOPI   = (int*)(SM + 2432);    // [200]
  int*   IIDX   = TOPI + 256;           // [200]

  auto cdiv = [](size_t a, size_t b) { return (unsigned)((a + b - 1) / b); };

  // embed + CNN
  k_embed<<<cdiv(Msz * 128, 256), 256, 0, stream>>>(x, F(1), F(2), XEMB);
  k_conv1<<<cdiv((size_t)BATCH * 64 * LSEQ, 256), 256, 0, stream>>>(
      x, F(3), F(4), F(5), F(6), CP0);
  dim3 gc(LSEQ / TM, 128 / TN, BATCH);
  k_conv_wmma<<<gc, 256, 0, stream>>>(CP0, F(7),  F(8),  F(9),  F(10), CP1, 64, 128, 0);
  k_conv_wmma<<<gc, 256, 0, stream>>>(CP1, F(11), F(12), F(13), F(14), CP0, 128, 128, 0);
  k_conv_wmma<<<gc, 256, 0, stream>>>(CP0, F(15), F(16), F(17), F(18), XC, 128, 128, 1);

  // x_combined = x_embed + x_cnn @ W_proj + b_proj; then + positional encoding
  gemm(XC, 128, F(19), 128, F(20), XEMB, XM, 128, MROWS, 128, 128, 0, stream);
  k_posenc<<<cdiv(Msz * 128, 256), 256, 0, stream>>>(XM);

  // BiMamba blocks. Backward direction: same token-pointwise GEMMs; only the
  // depthwise conv window and scan direction reverse (flip-free).
  for (int blk = 0; blk < 2; ++blk) {
    int base = blk ? 45 : 21;
    k_layernorm<<<cdiv(Msz, 256), 256, 0, stream>>>(XM, F(base), F(base + 1), XN);
    for (int dir = 0; dir < 2; ++dir) {
      int pb = base + 2 + dir * 11;  // W_in,b_in,conv_w,conv_b,W_xproj,W_dt,b_dt,A_log,D,W_out,b_out
      gemm(XN, 128, F(pb + 0), 512, F(pb + 1), nullptr, XZ, 512, MROWS, 512, 128, 0, stream);
      k_dwconv<<<cdiv(Msz * 256, 256), 256, 0, stream>>>(XZ, F(pb + 2), F(pb + 3), XA, dir);
      gemm(XA, 256, F(pb + 4), 40, nullptr, nullptr, DBC, 40, MROWS, 40, 256, 0, stream);
      gemm(DBC, 40, F(pb + 5), 256, F(pb + 6), nullptr, DTb, 256, MROWS, 256, 8, 3, stream);
      k_scan<<<BATCH, 256, 0, stream>>>(DTb, XA, DBC, F(pb + 7), YSb, dir);
      k_ymix<<<cdiv(Msz * 256, 256), 256, 0, stream>>>(YSb, XA, XZ, F(pb + 8));
      gemm(YSb, 256, F(pb + 9), 128, F(pb + 10), XM, XM, 128, MROWS, 128, 256, 0, stream);
    }
  }

  // main-effect path
  gemm(XM, 128, F(69), 64, F(70), nullptr, MEF, 64, MROWS, 64, 128, 1, stream);
  gemm(MEF, 64, F(71), 16, F(72), nullptr, MEF16, 16, MROWS, 16, 64, 0, stream);
  k_wmean<<<BATCH * 16, 256, 0, stream>>>(MEF16, F(87), XMW);
  k_colmean<<<BATCH * 128, 256, 0, stream>>>(XC, CMEAN);
  k_colmean<<<BATCH * 128, 256, 0, stream>>>(XM, FMEAN);

  // d_out: pred[0..8) main_weights[8..8200) epi_pairs[8200..8600) epi_scores[8600..8800)
  k_top200<<<LSEQ / 256, 256, 0, stream>>>(F(87), TOPI, out + 8);
  k_pairs<<<200, 256, 0, stream>>>(XM, TOPI, seli, pj, F(75), F(76), SCORES, IIDX);
  k_rank200<<<1, 256, 0, stream>>>(SCORES, IIDX, pj, out + 8600, (int*)(out + 8200));
  k_head<<<1, 32, 0, stream>>>(CMEAN, FMEAN, XMW, F(73), F(74), F(77), F(78),
                               F(79), F(80), F(81), F(82), F(83), F(84),
                               F(85), F(86), out);
}
